// TGCN_16896401342875
// MI455X (gfx1250) — matrix operand order repaired and checked by reading, not compiled, not run on hardware
//
#include <hip/hip_runtime.h>
#include <hip/hip_bf16.h>

// ---------------------------------------------------------------------------
// Problem constants (from the reference)
// ---------------------------------------------------------------------------
#define BSZ    64
#define SSEQ   80
#define NNODE  75
#define FIN    3
#define HDIM   128
#define KTAP   3
#define EEDGE  400
#define NCLS   100
#define RTOT   (BSZ * SSEQ * NNODE)      // 384000 flat rows [t*N+n]
#define SN     (SSEQ * NNODE)            // 6000 rows per batch element
#define NPART  1500                      // RTOT / 256 : BN partial blocks
#define BN_EPS 1e-5f

typedef __attribute__((ext_vector_type(16))) __bf16 v16bf;
typedef __attribute__((ext_vector_type(8)))  __bf16 v8bf;
typedef __attribute__((ext_vector_type(8)))  float  v8f;

// ---------------------------------------------------------------------------
// 1. Edge prep: build per-dst CSR + symmetric GCN norms (shared by all graphs)
// ---------------------------------------------------------------------------
__global__ void prep_edges(const int* __restrict__ ei,
                           int* __restrict__ csr_off, int* __restrict__ csr_src,
                           float* __restrict__ csr_w, float* __restrict__ self_w) {
  __shared__ int   cnt[NNODE + 1];
  __shared__ int   off[NNODE + 1];
  __shared__ float dinv[NNODE];
  const int t = threadIdx.x;
  if (t <= NNODE) cnt[t] = 0;
  __syncthreads();
  for (int e = t; e < EEDGE; e += blockDim.x) atomicAdd(&cnt[ei[EEDGE + e]], 1);
  __syncthreads();
  if (t == 0) {
    int s = 0;
    for (int i = 0; i < NNODE; ++i) { off[i] = s; s += cnt[i]; }
    off[NNODE] = s;
  }
  __syncthreads();
  if (t < NNODE) {
    const float deg = (float)(cnt[t] + 1);       // +1 self loop
    const float di  = rsqrtf(deg);
    dinv[t] = di;
    self_w[t] = di * di;
  }
  if (t <= NNODE) csr_off[t] = off[t];
  __syncthreads();
  if (t < NNODE) cnt[t] = 0;                     // reuse as cursor
  __syncthreads();
  for (int e = t; e < EEDGE; e += blockDim.x) {
    const int s = ei[e], d = ei[EEDGE + e];
    const int pos = off[d] + atomicAdd(&cnt[d], 1);
    csr_src[pos] = s;
    csr_w[pos]   = dinv[s] * dinv[d];
  }
}

// ---------------------------------------------------------------------------
// 2. Weight prep: bf16, transposed to [co][ci] so the B fragment's 16-wide K
//    run is one contiguous 32-byte load per lane.
// ---------------------------------------------------------------------------
__global__ void prep_weights(const float* __restrict__ W2,  const float* __restrict__ tw1,
                             const float* __restrict__ tw2, __bf16* __restrict__ W2t,
                             __bf16* __restrict__ tw1t, __bf16* __restrict__ tw2t) {
  const int co = blockIdx.x, ci = threadIdx.x;
  W2t[co * HDIM + ci] = (__bf16)W2[ci * HDIM + co];              // W2:[ci][co]
#pragma unroll
  for (int k = 0; k < KTAP; ++k) {                               // tw:[co][ci][k]
    tw1t[(k * HDIM + co) * HDIM + ci] = (__bf16)tw1[(co * HDIM + ci) * KTAP + k];
    tw2t[(k * HDIM + co) * HDIM + ci] = (__bf16)tw2[(co * HDIM + ci) * KTAP + k];
  }
}

// ---------------------------------------------------------------------------
// 3. Layer-1 feature transform (K = 3, pure VALU): y = x @ W1  (f32)
// ---------------------------------------------------------------------------
__global__ void xw1_kernel(const float* __restrict__ x, const float* __restrict__ W1,
                           float* __restrict__ Y) {
  const int r = blockIdx.x, c = threadIdx.x;
  const float* xr = x + (size_t)r * FIN;
  Y[(size_t)r * HDIM + c] =
      xr[0] * W1[c] + xr[1] * W1[HDIM + c] + xr[2] * W1[2 * HDIM + c];
}

// ---------------------------------------------------------------------------
// 4. GCN aggregation: gather over shared CSR, + self loop, + bias -> bf16
// ---------------------------------------------------------------------------
__global__ void gcn_agg(const float* __restrict__ XW, const float* __restrict__ bias,
                        const int* __restrict__ csr_off, const int* __restrict__ csr_src,
                        const float* __restrict__ csr_w, const float* __restrict__ self_w,
                        __bf16* __restrict__ Hout) {
  const int r = blockIdx.x, c = threadIdx.x;
  const int t = r / NNODE, n = r - t * NNODE;
  const size_t base = (size_t)t * NNODE * HDIM;
  float acc = self_w[n] * XW[(size_t)r * HDIM + c];
  const int e0 = csr_off[n], e1 = csr_off[n + 1];
  for (int e = e0; e < e1; ++e)
    acc += csr_w[e] * XW[base + (size_t)csr_src[e] * HDIM + c];
  Hout[(size_t)r * HDIM + c] = (__bf16)(acc + bias[c]);
}

// ---------------------------------------------------------------------------
// 5. WMMA matmul kernel. Computes Y[r,co] = sum_tap sum_ci
//       G[r + (tap - taps/2)*shiftStride, ci] * Wt[tap][co][ci]  (+ bias)
//    taps=1,shift=0  -> plain GEMM (GCN2 transform)
//    taps=3,shift=N  -> temporal Conv1d (pad=1) over the batch_seq axis,
//                       since t±1 at fixed node == flat row ±N.
//    One 256-thread block = 8 waves; wave w owns 16x16 output tile (r0, 16w).
//    bf16 operands fit in L2 (98 MB < 192 MB) -> direct global fragment loads.
// ---------------------------------------------------------------------------
__global__ __launch_bounds__(256) void wmma_mm(
    const __bf16* __restrict__ G, const __bf16* __restrict__ Wt,
    const float* __restrict__ bias, float* __restrict__ Y,
    int taps, int shiftStride) {
  const int lane = threadIdx.x & 31;
  const int wave = threadIdx.x >> 5;          // n-tile 0..7
  const int half = lane >> 4;                 // ISA: lane halves split K/M
  const int mn   = lane & 15;                 // A row / B col / C col
  const int r0   = blockIdx.x * 16;
  const int co0  = wave * 16;

  v8f acc = {};
  for (int tap = 0; tap < taps; ++tap) {
    const int row   = r0 + mn + (tap - (taps >> 1)) * shiftStride;
    const bool valid = (row >= 0) && (row < RTOT);   // zero padding at t edges
    const __bf16* Wtap = Wt + (size_t)tap * HDIM * HDIM;
#pragma unroll
    for (int kb = 0; kb < 4; ++kb) {          // 128 = 4 x K32
      // A fragment 16x32: lanes 0-15 hold K = kb*32 + {0..15} of row mn,
      // lanes 16-31 hold K = kb*32 + {16..31}  (matches B's contiguous K order).
      v16bf a = {};
      if (valid) {
        const __bf16* pa = G + (size_t)row * HDIM + kb * 32 + half * 16;
        const v8bf lo = *(const v8bf*)(pa);        // 16B load
        const v8bf hi = *(const v8bf*)(pa + 8);    // 16B load
        a = __builtin_shufflevector(lo, hi, 0, 1, 2, 3, 4, 5, 6, 7,
                                    8, 9, 10, 11, 12, 13, 14, 15);
      }
      // B fragment 32x16: lane holds 16 consecutive K (half*16) of col mn.
      const __bf16* pb = Wtap + (size_t)(co0 + mn) * HDIM + kb * 32 + half * 16;
      const v16bf b = *(const v16bf*)pb;           // 32B load
      acc = __builtin_amdgcn_wmma_f32_16x16x32_bf16(
          false, a, false, b, (short)0, acc, false, false);
    }
  }
  const float bia = bias ? bias[co0 + mn] : 0.0f;
#pragma unroll
  for (int v = 0; v < 8; ++v) {               // C/D: VGPR v -> row half*8+v
    const int orow = r0 + half * 8 + v;
    Y[(size_t)orow * HDIM + co0 + mn] = acc[v] + bia;
  }
}

// ---------------------------------------------------------------------------
// 6. BatchNorm stats: deterministic fixed-order two-stage reduction (no
//    float atomics -> graph replays are bitwise stable).
// ---------------------------------------------------------------------------
__global__ void bn_partial(const float* __restrict__ Y, float* __restrict__ part) {
  const int c = threadIdx.x;
  const size_t g = blockIdx.x;
  const float* base = Y + g * 256 * HDIM;
  float s = 0.f, s2 = 0.f;
  for (int i = 0; i < 256; ++i) {
    const float v = base[(size_t)i * HDIM + c];
    s += v; s2 += v * v;
  }
  part[(g * 2 + 0) * HDIM + c] = s;
  part[(g * 2 + 1) * HDIM + c] = s2;
}

__global__ void bn_finalize(const float* __restrict__ part, float* __restrict__ stats) {
  const int c = threadIdx.x;
  float s = 0.f, s2 = 0.f;
  for (int g = 0; g < NPART; ++g) {
    s  += part[(size_t)(g * 2 + 0) * HDIM + c];
    s2 += part[(size_t)(g * 2 + 1) * HDIM + c];
  }
  stats[c] = s;
  stats[HDIM + c] = s2;
}

// ---------------------------------------------------------------------------
// 7. BN apply + ReLU -> bf16 for next WMMA stage (optional f32 writeback for
//    the final pooling precision).
// ---------------------------------------------------------------------------
__global__ void bn_apply(const float* __restrict__ Y, const float* __restrict__ stats,
                         const float* __restrict__ gamma, const float* __restrict__ beta,
                         __bf16* __restrict__ Hout, float* __restrict__ Ywb) {
  const int c = threadIdx.x;
  const size_t r = blockIdx.x;
  const float invR = 1.0f / (float)RTOT;
  const float mu  = stats[c] * invR;
  const float var = stats[HDIM + c] * invR - mu * mu;     // population var
  const float sc  = gamma[c] * rsqrtf(var + BN_EPS);
  const float sh  = beta[c] - mu * sc;
  float v = Y[r * HDIM + c] * sc + sh;
  v = v > 0.f ? v : 0.f;                                  // ReLU
  Hout[r * HDIM + c] = (__bf16)v;
  if (Ywb) Ywb[r * HDIM + c] = v;
}

// ---------------------------------------------------------------------------
// 8. Mean over nodes then seq == mean over SN rows per batch element
// ---------------------------------------------------------------------------
__global__ void pool_kernel(const float* __restrict__ Y, float* __restrict__ pooled) {
  const int b = blockIdx.x, c = threadIdx.x;
  const float* base = Y + (size_t)b * SN * HDIM;
  float s = 0.f;
  for (int i = 0; i < SN; ++i) s += base[(size_t)i * HDIM + c];
  pooled[b * HDIM + c] = s * (1.0f / (float)SN);
}

// ---------------------------------------------------------------------------
// 9. FC head: 128 -> 64 (ReLU) -> 100
// ---------------------------------------------------------------------------
__global__ void head_kernel(const float* __restrict__ pooled,
                            const float* __restrict__ fc1w, const float* __restrict__ fc1b,
                            const float* __restrict__ fc2w, const float* __restrict__ fc2b,
                            float* __restrict__ out) {
  __shared__ float pb[HDIM];
  __shared__ float h1[HDIM / 2];
  const int b = blockIdx.x, t = threadIdx.x;
  pb[t] = pooled[b * HDIM + t];
  __syncthreads();
  if (t < HDIM / 2) {
    float s = fc1b[t];
    for (int i = 0; i < HDIM; ++i) s += pb[i] * fc1w[i * (HDIM / 2) + t];
    h1[t] = s > 0.f ? s : 0.f;
  }
  __syncthreads();
  if (t < NCLS) {
    float s = fc2b[t];
    for (int j = 0; j < HDIM / 2; ++j) s += h1[j] * fc2w[j * NCLS + t];
    out[b * NCLS + t] = s;
  }
}

// ---------------------------------------------------------------------------
// Launcher
// ---------------------------------------------------------------------------
extern "C" void kernel_launch(void* const* d_in, const int* in_sizes, int n_in,
                              void* d_out, int out_size, void* d_ws, size_t ws_size,
                              hipStream_t stream) {
  (void)in_sizes; (void)n_in; (void)out_size; (void)ws_size;
  const float* x    = (const float*)d_in[0];
  const int*   ei   = (const int*)  d_in[1];
  const float* W1   = (const float*)d_in[2];
  const float* b1   = (const float*)d_in[3];
  const float* tw1  = (const float*)d_in[4];
  const float* tb1  = (const float*)d_in[5];
  const float* g1   = (const float*)d_in[6];
  const float* be1  = (const float*)d_in[7];
  const float* W2   = (const float*)d_in[8];
  const float* b2   = (const float*)d_in[9];
  const float* tw2  = (const float*)d_in[10];
  const float* tb2  = (const float*)d_in[11];
  const float* g2   = (const float*)d_in[12];
  const float* be2  = (const float*)d_in[13];
  const float* fc1w = (const float*)d_in[14];
  const float* fc1b = (const float*)d_in[15];
  const float* fc2w = (const float*)d_in[16];
  const float* fc2b = (const float*)d_in[17];
  float* out = (float*)d_out;

  // Workspace carve-up (all offsets 256B aligned). Big buffers:
  //   Hbf: bf16 activations [R,128]  (98.3 MB, fits L2)
  //   Yf : f32  staging      [R,128]  (xw / conv-out, reused)
  char* ws = (char*)d_ws;
  __bf16* Hbf = (__bf16*)ws;                                   // 98,304,000 B
  float*  Yf  = (float*)(ws + 98304000);                       // 196,608,000 B
  char* sm = ws + 294912000;
  __bf16* W2t     = (__bf16*)sm;  sm += 32768;                 // 128x128 bf16
  __bf16* tw1t    = (__bf16*)sm;  sm += 98304;                 // 3x128x128 bf16
  __bf16* tw2t    = (__bf16*)sm;  sm += 98304;
  int*    csr_off = (int*)sm;     sm += 512;
  int*    csr_src = (int*)sm;     sm += 2048;
  float*  csr_w   = (float*)sm;   sm += 2048;
  float*  self_w  = (float*)sm;   sm += 512;
  float*  stats   = (float*)sm;   sm += 1024;                  // sum, sumsq
  float*  pooled  = (float*)sm;   sm += 32768;                 // 64x128
  float*  part    = (float*)sm;   sm += 6144000;               // 1500x2x128 f32

  const int MTILES = RTOT / 16;    // 24000

  // ---- prep ----
  prep_edges  <<<1,   128, 0, stream>>>(ei, csr_off, csr_src, csr_w, self_w);
  prep_weights<<<128, 128, 0, stream>>>(W2, tw1, tw2, W2t, tw1t, tw2t);

  // ---- layer 1 ----
  xw1_kernel<<<RTOT, 128, 0, stream>>>(x, W1, Yf);
  gcn_agg   <<<RTOT, 128, 0, stream>>>(Yf, b1, csr_off, csr_src, csr_w, self_w, Hbf);
  wmma_mm   <<<MTILES, 256, 0, stream>>>(Hbf, tw1t, tb1, Yf, KTAP, NNODE);   // conv1
  bn_partial<<<NPART, 128, 0, stream>>>(Yf, part);
  bn_finalize<<<1,   128, 0, stream>>>(part, stats);
  bn_apply  <<<RTOT, 128, 0, stream>>>(Yf, stats, g1, be1, Hbf, nullptr);

  // ---- layer 2 ----
  wmma_mm   <<<MTILES, 256, 0, stream>>>(Hbf, W2t, nullptr, Yf, 1, 0);       // xw2 GEMM
  gcn_agg   <<<RTOT, 128, 0, stream>>>(Yf, b2, csr_off, csr_src, csr_w, self_w, Hbf);
  wmma_mm   <<<MTILES, 256, 0, stream>>>(Hbf, tw2t, tb2, Yf, KTAP, NNODE);   // conv2
  bn_partial<<<NPART, 128, 0, stream>>>(Yf, part);
  bn_finalize<<<1,   128, 0, stream>>>(part, stats);
  bn_apply  <<<RTOT, 128, 0, stream>>>(Yf, stats, g2, be2, Hbf, Yf);         // f32 writeback

  // ---- pool + head ----
  pool_kernel<<<BSZ, 128, 0, stream>>>(Yf, pooled);
  head_kernel<<<BSZ, 128, 0, stream>>>(pooled, fc1w, fc1b, fc2w, fc2b, out);
}
